// Model_88905823027823
// MI455X (gfx1250) — matrix operand order repaired and checked by reading, not compiled, not run on hardware
//
#include <hip/hip_runtime.h>
#include <hip/hip_bf16.h>

// ---------------------------------------------------------------------------
// GINE bipartite GNN for MI455X (gfx1250, wave32, WMMA).
// All GEMMs (edge_attr@W_edge, node@W_nn, classifier z@W1) run on
// v_wmma_f32_16x16x32_bf16 with f32 accumulation; gather + bias + relu +
// scatter (global_atomic_add_f32) are fused into the WMMA epilogues.
// Full-tile fast paths keep the hot loops branch-free.
// ---------------------------------------------------------------------------

#define H 128
#define ED 32

typedef __attribute__((ext_vector_type(16))) __bf16 v16bf;
typedef __attribute__((ext_vector_type(8)))  float  v8f;

__device__ __forceinline__ unsigned short bf16_bits(float f) {
    union { float f; unsigned u; } in; in.f = f;
    unsigned u = in.u;
    unsigned r = u + 0x7FFFu + ((u >> 16) & 1u);   // round-to-nearest-even
    return (unsigned short)(r >> 16);
}

__device__ __forceinline__ __bf16 to_bf16(float f) {
    union { unsigned short s; __bf16 b; } o;
    o.s = bf16_bits(f);
    return o.b;
}

// B-matrix (32x16, 16-bit) per-lane K index:
// lanes 0-15: j0..7->K j, j8..15->K 16+(j-8) ; lanes 16-31: K += 8
__device__ __forceinline__ int kposB(int lane, int j) {
    return (j < 8 ? j : j + 8) + ((lane & 16) ? 8 : 0);
}

__device__ __forceinline__ v8f wmma_bf16(v16bf a, v16bf b, v8f c) {
    return __builtin_amdgcn_wmma_f32_16x16x32_bf16(
        /*neg_a=*/false, a, /*neg_b=*/false, b,
        /*c_mod=*/(short)0, c, /*reuse_a=*/false, /*reuse_b=*/false);
}

// Build one 16x32 bf16 A-fragment from a 32B-aligned f32 row pointer.
// A layout (ISA 7.12.2): lanes0-15 j0..7->K(b..b+7), j8..15->K(b+16..b+23); hi half b+=8.
__device__ __forceinline__ v16bf load_afrag(const float* __restrict__ p) {
    float4 f0 = *(const float4*)(p + 0);
    float4 f1 = *(const float4*)(p + 4);
    float4 f2 = *(const float4*)(p + 16);
    float4 f3 = *(const float4*)(p + 20);
    v16bf a;
    a[0]=to_bf16(f0.x); a[1]=to_bf16(f0.y); a[2]=to_bf16(f0.z); a[3]=to_bf16(f0.w);
    a[4]=to_bf16(f1.x); a[5]=to_bf16(f1.y); a[6]=to_bf16(f1.z); a[7]=to_bf16(f1.w);
    a[8]=to_bf16(f2.x); a[9]=to_bf16(f2.y); a[10]=to_bf16(f2.z); a[11]=to_bf16(f2.w);
    a[12]=to_bf16(f3.x); a[13]=to_bf16(f3.y); a[14]=to_bf16(f3.z); a[15]=to_bf16(f3.w);
    return a;
}

__device__ __forceinline__ v16bf zero_afrag() {
    v16bf a;
#pragma unroll
    for (int j = 0; j < 16; ++j) a[j] = to_bf16(0.0f);
    return a;
}

// ---------------------------------------------------------------------------
// init: x[i,:] = emb[idx[i],:]
// ---------------------------------------------------------------------------
__global__ void __launch_bounds__(256)
gather_rows_kernel(const float* __restrict__ emb, const int* __restrict__ idx,
                   float* __restrict__ out, int nrows) {
    long i = (long)blockIdx.x * 256 + threadIdx.x;
    long total = (long)nrows * (H / 4);
    if (i >= total) return;
    long row = i >> 5;            // 32 float4 per row
    int  c   = (int)(i & 31) * 4;
    long srow = idx[row];
    *(float4*)(out + row * H + c) = *(const float4*)(emb + srow * H + c);
}

// ---------------------------------------------------------------------------
// Edge kernel body: per wave, 16 edges x full H.
// ---------------------------------------------------------------------------
template<bool FULL>
__device__ __forceinline__ void edge_body(
    const float* __restrict__ edge_attr,
    const int*   __restrict__ src,
    const int*   __restrict__ dst,
    const float* __restrict__ x_src,
    const float* __restrict__ b_edge,
    float*       __restrict__ agg,
    const unsigned short* __restrict__ ldsW,
    long e0, int lane, int nE)
{
    const int  m  = lane & 15;
    const bool hi = (lane & 16) != 0;

    v16bf a;
    {
        long e = e0 + m;
        if (FULL || e < nE) a = load_afrag(edge_attr + e * ED + (hi ? 8 : 0));
        else                a = zero_afrag();
    }

    // row pointers for the 8 D-rows this lane owns (C/D layout: m = mb+j)
    const int mb = hi ? 8 : 0;
    const float* sp[8];
    float*       dp[8];
#pragma unroll
    for (int j = 0; j < 8; ++j) {
        long e = e0 + mb + j;
        if (FULL || e < nE) {
            sp[j] = x_src + (long)src[e] * H;
            dp[j] = agg   + (long)dst[e] * H;
        } else {
            sp[j] = x_src;
            dp[j] = nullptr;
        }
    }

    const int ncol = lane & 15;
#pragma unroll
    for (int t = 0; t < 8; ++t) {
        v16bf b = *reinterpret_cast<const v16bf*>(&ldsW[(t * 32 + lane) * 16]);
        v8f c = {0.f,0.f,0.f,0.f,0.f,0.f,0.f,0.f};
        c = wmma_bf16(a, b, c);
        const int h = t * 16 + ncol;
        const float be = b_edge[h];
        if (FULL) {
            float xs[8];
#pragma unroll
            for (int j = 0; j < 8; ++j) xs[j] = sp[j][h];        // batched gathers
#pragma unroll
            for (int j = 0; j < 8; ++j) {
                float msg = c[j] + xs[j] + be;
                msg = msg > 0.0f ? msg : 0.0f;
                atomicAdd(dp[j] + h, msg);
            }
        } else {
#pragma unroll
            for (int j = 0; j < 8; ++j) {
                if (dp[j]) {
                    float msg = c[j] + sp[j][h] + be;
                    msg = msg > 0.0f ? msg : 0.0f;
                    atomicAdd(dp[j] + h, msg);
                }
            }
        }
    }
}

__global__ void __launch_bounds__(256)
edge_gine_kernel(const float* __restrict__ edge_attr,
                 const int*   __restrict__ src,
                 const int*   __restrict__ dst,
                 const float* __restrict__ x_src,
                 const float* __restrict__ W_edge,
                 const float* __restrict__ b_edge,
                 float*       __restrict__ agg,
                 int nE) {
    __shared__ __align__(32) unsigned short ldsW[8 * 32 * 16];  // 8 KB
    const int tid = threadIdx.x;
    {   // stage W_edge (32x128) as 8 N-tiles of B fragments
        int t = tid >> 5, lane2 = tid & 31;
        int n = (t << 4) + (lane2 & 15);
        int base = tid * 16;
#pragma unroll
        for (int j = 0; j < 16; ++j)
            ldsW[base + j] = bf16_bits(W_edge[kposB(lane2, j) * H + n]);
    }
    __syncthreads();

    const int wave = tid >> 5, lane = tid & 31;
    const long e0 = (long)blockIdx.x * 128 + wave * 16;
    if (e0 >= nE) return;

    // prefetch next block's edge_attr slice into cache (global_prefetch_b8)
    {
        long pe = e0 + 128 + (lane & 15);
        if (pe < nE) __builtin_prefetch(edge_attr + pe * ED, 0, 1);
    }

    if (e0 + 16 <= nE)
        edge_body<true >(edge_attr, src, dst, x_src, b_edge, agg, ldsW, e0, lane, nE);
    else
        edge_body<false>(edge_attr, src, dst, x_src, b_edge, agg, ldsW, e0, lane, nE);
}

// ---------------------------------------------------------------------------
// Node update body: out = ((1+eps)*x + agg) @ W_nn + b_nn  [+relu]
// ---------------------------------------------------------------------------
template<bool FULL>
__device__ __forceinline__ void node_body(
    const float* __restrict__ x,
    const float* __restrict__ agg,
    const float* __restrict__ b_nn,
    float*       __restrict__ out,
    const unsigned short* __restrict__ ldsW,
    long r0, int lane, int N, float ep1, int do_relu)
{
    const int  m  = lane & 15;
    const bool hi = (lane & 16) != 0;
    const long r  = r0 + m;

    v16bf afr[4];
    if (FULL || r < N) {
        const float* xp = x   + r * H;
        const float* gp = agg + r * H;
#pragma unroll
        for (int kb = 0; kb < 4; ++kb) {
            int base = kb * 32 + (hi ? 8 : 0);
            float4 x0 = *(const float4*)(xp + base);
            float4 x1 = *(const float4*)(xp + base + 4);
            float4 x2 = *(const float4*)(xp + base + 16);
            float4 x3 = *(const float4*)(xp + base + 20);
            float4 g0 = *(const float4*)(gp + base);
            float4 g1 = *(const float4*)(gp + base + 4);
            float4 g2 = *(const float4*)(gp + base + 16);
            float4 g3 = *(const float4*)(gp + base + 20);
            afr[kb][0]=to_bf16(ep1*x0.x+g0.x); afr[kb][1]=to_bf16(ep1*x0.y+g0.y);
            afr[kb][2]=to_bf16(ep1*x0.z+g0.z); afr[kb][3]=to_bf16(ep1*x0.w+g0.w);
            afr[kb][4]=to_bf16(ep1*x1.x+g1.x); afr[kb][5]=to_bf16(ep1*x1.y+g1.y);
            afr[kb][6]=to_bf16(ep1*x1.z+g1.z); afr[kb][7]=to_bf16(ep1*x1.w+g1.w);
            afr[kb][8]=to_bf16(ep1*x2.x+g2.x); afr[kb][9]=to_bf16(ep1*x2.y+g2.y);
            afr[kb][10]=to_bf16(ep1*x2.z+g2.z); afr[kb][11]=to_bf16(ep1*x2.w+g2.w);
            afr[kb][12]=to_bf16(ep1*x3.x+g3.x); afr[kb][13]=to_bf16(ep1*x3.y+g3.y);
            afr[kb][14]=to_bf16(ep1*x3.z+g3.z); afr[kb][15]=to_bf16(ep1*x3.w+g3.w);
        }
    } else {
#pragma unroll
        for (int kb = 0; kb < 4; ++kb) afr[kb] = zero_afrag();
    }

    const int mb = hi ? 8 : 0;
    const int ncol = lane & 15;
#pragma unroll
    for (int t = 0; t < 8; ++t) {
        v8f c = {0.f,0.f,0.f,0.f,0.f,0.f,0.f,0.f};
#pragma unroll
        for (int kb = 0; kb < 4; ++kb) {
            v16bf b = *reinterpret_cast<const v16bf*>(&ldsW[((t * 4 + kb) * 32 + lane) * 16]);
            c = wmma_bf16(afr[kb], b, c);
        }
        const int h = t * 16 + ncol;
        const float bn = b_nn[h];
#pragma unroll
        for (int j = 0; j < 8; ++j) {
            long rr = r0 + mb + j;
            if (FULL || rr < N) {
                float v = c[j] + bn;
                if (do_relu) v = v > 0.0f ? v : 0.0f;
                out[rr * H + h] = v;
            }
        }
    }
}

__global__ void __launch_bounds__(256)
node_update_kernel(const float* __restrict__ x,
                   const float* __restrict__ agg,
                   const float* __restrict__ W_nn,
                   const float* __restrict__ b_nn,
                   const float* __restrict__ eps_p,
                   float*       __restrict__ out,
                   int N, int do_relu) {
    __shared__ __align__(32) unsigned short ldsW[8 * 4 * 32 * 16];  // 32 KB
    const int tid = threadIdx.x;
    for (int f = tid; f < 1024; f += 256) {
        int ntile = f >> 7;
        int rem   = f & 127;
        int kb    = rem >> 5;
        int lane2 = rem & 31;
        int n     = ntile * 16 + (lane2 & 15);
        int base  = f * 16;
#pragma unroll
        for (int j = 0; j < 16; ++j)
            ldsW[base + j] = bf16_bits(W_nn[(kb * 32 + kposB(lane2, j)) * H + n]);
    }
    __syncthreads();

    const float ep1 = 1.0f + eps_p[0];
    const int wave = tid >> 5, lane = tid & 31;
    const long r0 = (long)blockIdx.x * 128 + wave * 16;
    if (r0 >= N) return;

    if (r0 + 16 <= N)
        node_body<true >(x, agg, b_nn, out, ldsW, r0, lane, N, ep1, do_relu);
    else
        node_body<false>(x, agg, b_nn, out, ldsW, r0, lane, N, ep1, do_relu);
}

// ---------------------------------------------------------------------------
// Classifier: z = [xu[ui], xb[bi]] (256) ; hrow = relu(z@W1+b1) (128)
// pred = hrow . W2 + b2.  W1 (256x128) staged in LDS (64 KB, fragment order).
// ---------------------------------------------------------------------------
__global__ void __launch_bounds__(256)
classifier_kernel(const int*   __restrict__ eli_u,
                  const int*   __restrict__ eli_b,
                  const float* __restrict__ xu,
                  const float* __restrict__ xb,
                  const float* __restrict__ W1,
                  const float* __restrict__ b1,
                  const float* __restrict__ W2,
                  const float* __restrict__ b2,
                  float*       __restrict__ pred,
                  int nEL) {
    __shared__ __align__(32) unsigned short ldsW[8 * 8 * 32 * 16];  // 64 KB
    const int tid = threadIdx.x;
    for (int f = tid; f < 2048; f += 256) {
        int ntile = f >> 8;
        int rem   = f & 255;
        int kb    = rem >> 5;
        int lane2 = rem & 31;
        int n     = ntile * 16 + (lane2 & 15);
        int base  = f * 16;
#pragma unroll
        for (int j = 0; j < 16; ++j)
            ldsW[base + j] = bf16_bits(W1[(kb * 32 + kposB(lane2, j)) * H + n]);
    }
    __syncthreads();

    const int wave = tid >> 5, lane = tid & 31;
    const long e0 = (long)blockIdx.x * 128 + wave * 16;
    if (e0 >= nEL) return;

    const int  m  = lane & 15;
    const bool hi = (lane & 16) != 0;
    long el = e0 + m;
    int ui = 0, bi = 0;
    if (el < nEL) { ui = eli_u[el]; bi = eli_b[el]; }   // clamped: garbage rows never written
    const float* up = xu + (long)ui * H;
    const float* bp = xb + (long)bi * H;

    v16bf afr[8];
#pragma unroll
    for (int kb = 0; kb < 8; ++kb) {
        const float* p = (kb < 4) ? up : bp;
        afr[kb] = load_afrag(p + (kb & 3) * 32 + (hi ? 8 : 0));
    }

    float acc[8] = {0.f,0.f,0.f,0.f,0.f,0.f,0.f,0.f};
    const int ncol = lane & 15;
#pragma unroll
    for (int t = 0; t < 8; ++t) {
        v8f c = {0.f,0.f,0.f,0.f,0.f,0.f,0.f,0.f};
#pragma unroll
        for (int kb = 0; kb < 8; ++kb) {
            v16bf b = *reinterpret_cast<const v16bf*>(&ldsW[((t * 8 + kb) * 32 + lane) * 16]);
            c = wmma_bf16(afr[kb], b, c);
        }
        const int h = t * 16 + ncol;
        const float bb = b1[h];
        const float w2 = W2[h];
#pragma unroll
        for (int j = 0; j < 8; ++j) {
            float v = c[j] + bb;
            v = v > 0.0f ? v : 0.0f;
            acc[j] += v * w2;
        }
    }
    // reduce each row's partial sums across the 16 lanes of each half-wave
#pragma unroll
    for (int j = 0; j < 8; ++j) {
        float v = acc[j];
        v += __shfl_xor(v, 1, 32);
        v += __shfl_xor(v, 2, 32);
        v += __shfl_xor(v, 4, 32);
        v += __shfl_xor(v, 8, 32);
        acc[j] = v;
    }
    const int mb = hi ? 8 : 0;
    if ((lane & 15) == 0) {
        const float b2v = b2[0];
#pragma unroll
        for (int j = 0; j < 8; ++j) {
            long ee = e0 + mb + j;
            if (ee < nEL) pred[ee] = acc[j] + b2v;
        }
    }
}

// ---------------------------------------------------------------------------
extern "C" void kernel_launch(void* const* d_in, const int* in_sizes, int n_in,
                              void* d_out, int out_size, void* d_ws, size_t ws_size,
                              hipStream_t stream) {
    (void)n_in; (void)out_size; (void)ws_size;
    const int NU = in_sizes[0];          // 200000
    const int NB = in_sizes[1];          // 100000
    const int E  = in_sizes[2] / 2;      // 1000000
    const int EL = in_sizes[6] / 2;      // 500000

    const int*   n_id_user = (const int*)  d_in[0];
    const int*   n_id_book = (const int*)  d_in[1];
    const int*   ei_ub     = (const int*)  d_in[2];   // [0,E)=user src, [E,2E)=book dst
    const float* ea_ub     = (const float*)d_in[3];
    const int*   ei_bu     = (const int*)  d_in[4];   // [0,E)=book src, [E,2E)=user dst
    const float* ea_bu     = (const float*)d_in[5];
    const int*   eli       = (const int*)  d_in[6];
    const float* user_emb  = (const float*)d_in[7];
    const float* book_emb  = (const float*)d_in[8];
    const float* W_edge    = (const float*)d_in[9];
    const float* b_edge    = (const float*)d_in[10];
    const float* W_nn      = (const float*)d_in[11];
    const float* b_nn      = (const float*)d_in[12];
    const float* eps       = (const float*)d_in[13];
    const float* W1        = (const float*)d_in[14];
    const float* b1        = (const float*)d_in[15];
    const float* W2        = (const float*)d_in[16];
    const float* b2        = (const float*)d_in[17];

    float* pred_out = (float*)d_out;
    float* xu_out   = pred_out + EL;
    float* xb_out   = xu_out + (size_t)NU * H;

    char*  ws  = (char*)d_ws;
    size_t szU = (size_t)NU * H * sizeof(float);
    size_t szB = (size_t)NB * H * sizeof(float);
    float* bufU0 = (float*)(ws);
    float* bufB0 = (float*)(ws + szU);
    float* bufU1 = (float*)(ws + szU + szB);
    float* bufB1 = (float*)(ws + 2 * szU + szB);

    const int eblocks = (E  + 127) / 128;
    const int ublocks = (NU + 127) / 128;
    const int bblocks = (NB + 127) / 128;
    const int lblocks = (EL + 127) / 128;

    // x_user = user_emb[n_id_user] ; x_book = book_emb[n_id_book]
    gather_rows_kernel<<<(NU * (H / 4) + 255) / 256, 256, 0, stream>>>(user_emb, n_id_user, bufU0, NU);
    gather_rows_kernel<<<(NB * (H / 4) + 255) / 256, 256, 0, stream>>>(book_emb, n_id_book, bufB0, NB);

    // ---- Layer 0 ----
    hipMemsetAsync(bufU1, 0, szU, stream);   // agg_user
    hipMemsetAsync(bufB1, 0, szB, stream);   // agg_book
    edge_gine_kernel<<<eblocks, 256, 0, stream>>>(ea_ub, ei_ub, ei_ub + E, bufU0, W_edge, b_edge, bufB1, E);
    edge_gine_kernel<<<eblocks, 256, 0, stream>>>(ea_bu, ei_bu, ei_bu + E, bufB0, W_edge, b_edge, bufU1, E);
    node_update_kernel<<<bblocks, 256, 0, stream>>>(bufB0, bufB1, W_nn, b_nn, eps, bufB1, NB, 1);
    node_update_kernel<<<ublocks, 256, 0, stream>>>(bufU0, bufU1, W_nn, b_nn, eps, bufU1, NU, 1);

    // ---- Layer 1 (final; no relu; write straight to d_out) ----
    hipMemsetAsync(bufU0, 0, szU, stream);   // agg_user
    hipMemsetAsync(bufB0, 0, szB, stream);   // agg_book
    edge_gine_kernel<<<eblocks, 256, 0, stream>>>(ea_ub, ei_ub, ei_ub + E, bufU1, W_edge, b_edge, bufB0, E);
    edge_gine_kernel<<<eblocks, 256, 0, stream>>>(ea_bu, ei_bu, ei_bu + E, bufB1, W_edge, b_edge, bufU0, E);
    node_update_kernel<<<bblocks, 256, 0, stream>>>(bufB1, bufB0, W_nn, b_nn, eps, xb_out, NB, 0);
    node_update_kernel<<<ublocks, 256, 0, stream>>>(bufU1, bufU0, W_nn, b_nn, eps, xu_out, NU, 0);

    // ---- Edge classifier ----
    classifier_kernel<<<lblocks, 256, 0, stream>>>(eli, eli + EL, xu_out, xb_out,
                                                   W1, b1, W2, b2, pred_out, EL);
}
